// GraphNgmAttentionLayer_3410204033349
// MI455X (gfx1250) — matrix operand order repaired and checked by reading, not compiled, not run on hardware
//
#include <hip/hip_runtime.h>
#include <hip/hip_bf16.h>

// ---------------------------------------------------------------------------
// GraphNgmAttentionLayer for MI455X (gfx1250, wave32).
//
// Roofline: GEMM h=XW is 1.64 GFLOP over ~51MB -> HBM-bound (~2.2us @ 23.3TB/s).
// Attention factorizes: e_nbr = (h@c2)[adj], so precompute hc1/hc2 per node.
// K_NBR == 32 == wave32: the softmax/cumsum/window/argmax row pipeline is pure
// cross-lane shuffles. Only SIZE=3 neighbors survive windowing, so the output
// gather is 3 h-rows/node from the 192MB L2 (h is 25.6MB, L2-resident).
//
// v2: X stripe staged to LDS as bf16 once per block (was: 8x redundant global
// loads + per-wave integer bf16 rounding). fp32->bf16 now via native cast so
// clang emits the packed hardware convert instead of bfe/add3 sequences.
// ---------------------------------------------------------------------------

#define N_NODES 50000
#define FDIM    128

typedef __attribute__((ext_vector_type(16))) __bf16  v16bf;
typedef __attribute__((ext_vector_type(8)))  __bf16  v8bf;
typedef __attribute__((ext_vector_type(8)))  float   v8f;

// ---------------------------------------------------------------------------
// Kernel 1: H = X * W via v_wmma_f32_16x16x32_bf16.
// Block = 256 threads = 8 waves; block owns a 16-row stripe of X and all 128
// output columns (wave w -> column tile w). W (128x128) staged transposed in
// LDS as bf16 (32KB); X stripe (16x128) staged in LDS as bf16 (4KB). Inner
// loop is ds_load-only.
// Fragment layouts used here (contiguous k, same order in A and B):
//   A(16x32 bf16): lane&15 = M row; half=lane>>4: elems0..15 = K(k0+half*16 .. +15)
//   B(32x16 bf16): lane&15 = N col; elems0..15 = K(k0+half*16 .. +15)
//   D(16x16 f32):  lane&15 = N col; elem r -> M row (r + half*8)
// ---------------------------------------------------------------------------
__global__ void k_gemm(const float* __restrict__ X, const float* __restrict__ W,
                       float* __restrict__ H) {
    __shared__ __bf16 wt[FDIM * FDIM];   // wt[n*128 + k]  (W transposed)
    __shared__ __bf16 xa[16 * FDIM];     // xa[r*128 + k]  (X stripe)
    const int t = threadIdx.x;
    const int rowBase = blockIdx.x * 16; // 50000/16 = 3125 blocks exactly

    // stage W (coalesced global read, transposed LDS write, one convert each)
    for (int i = t; i < FDIM * FDIM; i += 256) {
        const int k = i >> 7, n = i & 127;
        wt[n * FDIM + k] = (__bf16)W[i];
    }
    // stage the 16x128 X stripe (2048 elems, 8 per thread, coalesced)
    for (int i = t; i < 16 * FDIM; i += 256) {
        xa[i] = (__bf16)X[(size_t)rowBase * FDIM + i];
    }
    __syncthreads();

    const int wave = t >> 5, lane = t & 31;
    const int half = lane >> 4, mr = lane & 15;
    const int colBase = wave * 16;
    const int n = colBase + mr;

    v8f c = {};
    #pragma unroll
    for (int k0 = 0; k0 < FDIM; k0 += 32) {
        // ---- A fragment: two 16B LDS chunks, stitched ----
        const int ka = k0 + half * 16;
        v8bf alo = *(const v8bf*)(&xa[mr * FDIM + ka]);
        v8bf ahi = *(const v8bf*)(&xa[mr * FDIM + ka + 8]);
        v16bf a = __builtin_shufflevector(alo, ahi,
                                          0, 1, 2, 3, 4, 5, 6, 7,
                                          8, 9, 10, 11, 12, 13, 14, 15);
        // ---- B fragment: contiguous 32B LDS chunk ----
        const int kb = k0 + half * 16;
        v16bf b = *(const v16bf*)(&wt[n * FDIM + kb]);
        // D = A*B + C  (f32 accumulate)
        c = __builtin_amdgcn_wmma_f32_16x16x32_bf16(
                false, a, false, b, (short)0, c, false, false);
    }
    // ---- store D tile (rows of 64B, coalesced per half-wave) ----
    #pragma unroll
    for (int r = 0; r < 8; ++r) {
        const int m = rowBase + r + half * 8;
        H[(size_t)m * FDIM + n] = c[r];
    }
}

// ---------------------------------------------------------------------------
// Kernel 2: hc1[n] = h[n].c1 ; hc2[n] = h[n].c2. Wave per node, lane = 4 feats.
// ---------------------------------------------------------------------------
__global__ void k_proj(const float* __restrict__ H, const float* __restrict__ c1,
                       const float* __restrict__ c2, float* __restrict__ hc1,
                       float* __restrict__ hc2) {
    const int wid  = (blockIdx.x * blockDim.x + threadIdx.x) >> 5;
    const int lane = threadIdx.x & 31;
    if (wid >= N_NODES) return;
    float4 hv = ((const float4*)(H + (size_t)wid * FDIM))[lane];
    float4 a  = ((const float4*)c1)[lane];
    float4 b  = ((const float4*)c2)[lane];
    float d1 = hv.x * a.x + hv.y * a.y + hv.z * a.z + hv.w * a.w;
    float d2 = hv.x * b.x + hv.y * b.y + hv.z * b.z + hv.w * b.w;
    #pragma unroll
    for (int off = 16; off; off >>= 1) {
        d1 += __shfl_xor(d1, off, 32);
        d2 += __shfl_xor(d2, off, 32);
    }
    if (lane == 0) { hc1[wid] = d1; hc2[wid] = d2; }
}

// ---------------------------------------------------------------------------
// Kernel 3: per-node attention row (wave32 == K_NBR=32), sliding window,
// weighted gather of the 3 surviving neighbor rows, ELU, coalesced store.
// ---------------------------------------------------------------------------
__global__ void k_attn(const float* __restrict__ H, const float* __restrict__ hc1,
                       const float* __restrict__ hc2, const int* __restrict__ adj,
                       const int* __restrict__ deg, float* __restrict__ out) {
    const int wid  = (blockIdx.x * blockDim.x + threadIdx.x) >> 5;
    const int lane = threadIdx.x & 31;
    if (wid >= N_NODES) return;

    const int nbr = adj[(size_t)wid * 32 + lane];
    const int dg  = deg[wid];
    const bool valid = lane < dg;

    // leaky_relu(e_self + e_nbr), masked
    float e = hc1[wid] + hc2[nbr];
    e = (e > 0.f) ? e : 0.2f * e;
    e = valid ? e : -3.0e38f;

    // masked softmax across the wave
    float mx = e;
    #pragma unroll
    for (int off = 16; off; off >>= 1) mx = fmaxf(mx, __shfl_xor(mx, off, 32));
    float ex = valid ? __expf(e - mx) : 0.f;
    float sm = ex;
    #pragma unroll
    for (int off = 16; off; off >>= 1) sm += __shfl_xor(sm, off, 32);
    const float att = ex / sm;

    // inclusive prefix sum of att across lanes
    float cs = att;
    #pragma unroll
    for (int off = 1; off < 32; off <<= 1) {
        float tv = __shfl_up(cs, off, 32);
        if (lane >= off) cs += tv;
    }

    // window sum of width 3 ending at this lane
    float csm3 = __shfl_up(cs, 3, 32);
    float win = cs - ((lane >= 3) ? csm3 : 0.f);
    win = ((lane >= 2) && valid) ? win : -1.0f;

    // first-argmax across the wave: max-reduce then lowest set lane of ballot
    float wmax = win;
    #pragma unroll
    for (int off = 16; off; off >>= 1) wmax = fmaxf(wmax, __shfl_xor(wmax, off, 32));
    unsigned long long bal = __ballot(win == wmax);
    const int jstar = __ffsll(bal) - 1;

    // keep window [jstar-2, jstar], rescale by deg/SIZE
    const bool keep = (lane >= jstar - 2) && (lane <= jstar);
    const float af = keep ? att * ((float)dg * (1.0f / 3.0f)) : 0.f;

    // gather + weighted sum of the 3 kept neighbor rows (L2-resident H)
    float4 acc = {0.f, 0.f, 0.f, 0.f};
    const float4* h4 = (const float4*)H;
    #pragma unroll
    for (int s = 0; s < 3; ++s) {
        const int kk = (jstar - s) & 31;        // wrap safe: wrapped lane has af==0
        const float a = __shfl(af, kk, 32);
        const int   id = __shfl(nbr, kk, 32);
        float4 r = h4[(size_t)id * 32 + lane];
        acc.x += a * r.x; acc.y += a * r.y; acc.z += a * r.z; acc.w += a * r.w;
    }
    // ELU (alpha = 1)
    acc.x = (acc.x > 0.f) ? acc.x : __expf(acc.x) - 1.f;
    acc.y = (acc.y > 0.f) ? acc.y : __expf(acc.y) - 1.f;
    acc.z = (acc.z > 0.f) ? acc.z : __expf(acc.z) - 1.f;
    acc.w = (acc.w > 0.f) ? acc.w : __expf(acc.w) - 1.f;
    ((float4*)out)[(size_t)wid * 32 + lane] = acc;
}

// ---------------------------------------------------------------------------
extern "C" void kernel_launch(void* const* d_in, const int* in_sizes, int n_in,
                              void* d_out, int out_size, void* d_ws, size_t ws_size,
                              hipStream_t stream) {
    const float* X   = (const float*)d_in[0];   // input  [N,128] f32
    const float* W   = (const float*)d_in[1];   // W      [128,128] f32
    const float* c1  = (const float*)d_in[2];   // c1     [128,1] f32
    const float* c2  = (const float*)d_in[3];   // c2     [128,1] f32
    const int*   adj = (const int*)d_in[4];     // adj    [M,32] i32
    const int*   deg = (const int*)d_in[5];     // deg    [M]    i32
    float* out = (float*)d_out;                 // [M,128] f32

    // workspace: H (25.6MB) | hc1 (200KB) | hc2 (200KB)
    float* H   = (float*)d_ws;
    float* hc1 = H + (size_t)N_NODES * FDIM;
    float* hc2 = hc1 + N_NODES;

    // 50000/16 = 3125 row tiles; 8 waves/block cover all 8 column tiles
    k_gemm<<<N_NODES / 16, 256, 0, stream>>>(X, W, H);

    const int wavesPerBlock = 8;                // 256 threads = 8 wave32
    const int blocks = (N_NODES + wavesPerBlock - 1) / wavesPerBlock;
    k_proj<<<blocks, 256, 0, stream>>>(H, c1, c2, hc1, hc2);
    k_attn<<<blocks, 256, 0, stream>>>(H, hc1, hc2, adj, deg, out);
}
